// Head_67972152426928
// MI455X (gfx1250) — compile-verified, matrix operand order checked by reading, not run on hardware
//
#include <hip/hip_runtime.h>

typedef __bf16 bf16;
typedef unsigned int u32;
typedef __attribute__((ext_vector_type(16))) __bf16 v16bf;
typedef __attribute__((ext_vector_type(8)))  float  v8f;
typedef __attribute__((ext_vector_type(8)))  u32    v8u;

#define BN 512
#define TN 256
#define CN 384
#define HSN 64

union FragU { v8u u; v16bf b; };          // 32-byte reinterpret, same VGPRs
union PackU { u32 u; bf16 h[2]; };

static __device__ __forceinline__ u32 pack_bf16(float lo, float hi) {
    PackU p;
    p.h[0] = (bf16)lo;
    p.h[1] = (bf16)hi;
    return p.u;                            // lowers to v_cvt_pk_bf16_f32
}

// K-pair uint index for A/B fragment element v (ISA 7.12.2 16-bit layout)
#define KPIDX(kk, v, half) ((kk) * 16 + 8 * ((v) >> 2) + 4 * (half) + ((v) & 3))

// --------------------------------------------------------------------------
// Kernel 1: QKV projection.  [B*T, C] (f32) @ [C, HS] (f32) -> bf16 [B*T, HS]
// grid = (B*T/64, 3), block = 128 (4 waves). Each wave -> 16x64 output tile.
// --------------------------------------------------------------------------
__global__ __launch_bounds__(128) void qkv_proj_kernel(
    const float* __restrict__ x,
    const float* __restrict__ Wk,
    const float* __restrict__ Wq,
    const float* __restrict__ Wv,
    bf16* __restrict__ Kb, bf16* __restrict__ Qb, bf16* __restrict__ Vb)
{
    __shared__ u32 xs[64][CN / 2 + 4];     // 64x196 u32 = 50176 B (K-pair packed)
    __shared__ u32 wt[64][16 + 4];         // 64x20  u32 =  5120 B (transposed W chunk)

    const int tid = threadIdx.x;
    const float* W;
    bf16* O;
    if (blockIdx.y == 0)      { W = Wk; O = Kb; }
    else if (blockIdx.y == 1) { W = Wq; O = Qb; }
    else                      { W = Wv; O = Vb; }

    const long row0 = (long)blockIdx.x * 64;

    // stage x tile: float4 global loads -> packed bf16 pairs in LDS
    const float4* x4 = (const float4*)x;
    for (int i = tid; i < 64 * (CN / 4); i += 128) {       // 64 x 96 float4
        int r = i / (CN / 4), c = i % (CN / 4);
        float4 xv = x4[(row0 + r) * (CN / 4) + c];
        xs[r][2 * c]     = pack_bf16(xv.x, xv.y);
        xs[r][2 * c + 1] = pack_bf16(xv.z, xv.w);
    }

    const int wave = tid >> 5;
    const int lane = tid & 31;
    const int half = lane >> 4;
    const int lrow = lane & 15;

    v8f acc[4];
    #pragma unroll
    for (int nt = 0; nt < 4; ++nt)
        #pragma unroll
        for (int e = 0; e < 8; ++e) acc[nt][e] = 0.0f;

    for (int ks = 0; ks < CN / 32; ++ks) {                 // 12 K-steps of 32
        const int k0 = ks * 32;
        __syncthreads();                                   // wt reuse (also xs ready)
        // stage W chunk TRANSPOSED: wt[n][kpair]
        for (int i = tid; i < 64 * 16; i += 128) {
            int c = i & 63, kp = i >> 6;
            float w0 = W[(k0 + 2 * kp)     * HSN + c];
            float w1 = W[(k0 + 2 * kp + 1) * HSN + c];
            wt[c][kp] = pack_bf16(w0, w1);
        }
        __syncthreads();

        FragU a;
        #pragma unroll
        for (int v = 0; v < 8; ++v)
            a.u[v] = xs[wave * 16 + lrow][KPIDX(ks, v, half)];
        #pragma unroll
        for (int nt = 0; nt < 4; ++nt) {
            FragU b;
            #pragma unroll
            for (int v = 0; v < 8; ++v)
                b.u[v] = wt[nt * 16 + lrow][KPIDX(0, v, half)];
            acc[nt] = __builtin_amdgcn_wmma_f32_16x16x32_bf16(
                false, a.b, false, b.b, (short)0, acc[nt], false, false);
        }
    }

    #pragma unroll
    for (int nt = 0; nt < 4; ++nt)
        #pragma unroll
        for (int v = 0; v < 8; ++v) {
            long r = row0 + wave * 16 + v + 8 * half;
            O[r * HSN + nt * 16 + lrow] = (bf16)acc[nt][v];
        }
}

// --------------------------------------------------------------------------
// Kernel 2: causal flash attention over bf16 Q/K/V, fp32 softmax + output.
// grid = (T/64, B), block = 128 (4 waves). Each wave owns 16 query rows.
// Softmax runs in the log2 domain: one v_exp_f32 per exponential.
// --------------------------------------------------------------------------
__global__ __launch_bounds__(128) void flash_attn_kernel(
    const bf16* __restrict__ Qb, const bf16* __restrict__ Kb,
    const bf16* __restrict__ Vb, float* __restrict__ out)
{
    __shared__ u32 Qs[64][36];             // [row][kpair]           9216 B
    __shared__ u32 Ks[64][36];             // [key row][kpair]       9216 B
    __shared__ u32 Vt[64][36];             // TRANSPOSED: [col][row-pair]
    __shared__ u32 Ps[4][16][36];          // per-wave P staging

    const int tid  = threadIdx.x;
    const int wave = tid >> 5, lane = tid & 31;
    const int half = lane >> 4, lrow = lane & 15;
    const int b  = blockIdx.y;
    const int qt = blockIdx.x;
    const long base = (long)b * TN * HSN;
    // 384^-0.5 * log2(e): softmax computed with exp2
    const float scale2 = 0.07362222f;

    // stage Q tile: uint4 copies (rows are 128B-aligned)
    const uint4* Qb4 = (const uint4*)(Qb + base + (long)qt * 64 * HSN);
    for (int i = tid; i < 64 * 8; i += 128)
        *(uint4*)&Qs[i >> 3][(i & 7) * 4] = Qb4[i];
    __syncthreads();

    // Q A-fragments are j-invariant: build once.
    FragU aq[2];
    #pragma unroll
    for (int kk = 0; kk < 2; ++kk)
        #pragma unroll
        for (int v = 0; v < 8; ++v)
            aq[kk].u[v] = Qs[wave * 16 + lrow][KPIDX(kk, v, half)];

    v8f o[4];
    float mi[8], li[8];
    #pragma unroll
    for (int nt = 0; nt < 4; ++nt)
        #pragma unroll
        for (int e = 0; e < 8; ++e) o[nt][e] = 0.0f;
    #pragma unroll
    for (int v = 0; v < 8; ++v) { mi[v] = -__builtin_inff(); li[v] = 0.0f; }

    for (int j = 0; j <= qt; ++j) {
        __syncthreads();                   // prior iteration done with Ks/Vt
        const uint4* Kb4  = (const uint4*)(Kb + base + (long)j * 64 * HSN);
        const u32*   Vb32 = (const u32*)  (Vb + base + (long)j * 64 * HSN);
        for (int i = tid; i < 64 * 8; i += 128)
            *(uint4*)&Ks[i >> 3][(i & 7) * 4] = Kb4[i];
        // V staged transposed: 2x2 bf16 micro-transpose in registers, b32 stores
        for (int i = tid; i < 64 * 16; i += 128) {         // 1024 2x2 blocks
            int rp = i >> 5, cp = i & 31;                  // rows 2rp,2rp+1 cols 2cp,2cp+1
            u32 a = Vb32[(2 * rp)     * 32 + cp];
            u32 c = Vb32[(2 * rp + 1) * 32 + cp];
            Vt[2 * cp][rp]     = (a & 0xffffu)  | (c << 16);
            Vt[2 * cp + 1][rp] = (a >> 16)      | (c & 0xffff0000u);
        }
        if (j < qt) {                      // prefetch next tile (global_prefetch_b8)
            __builtin_prefetch(&Kb[base + (long)((j + 1) * 64 + (tid & 63)) * HSN], 0, 1);
            __builtin_prefetch(&Vb[base + (long)((j + 1) * 64 + (tid & 63)) * HSN], 0, 1);
        }
        __syncthreads();

        // S = Q @ K^T  (K rows read along feature dim -> contiguous pairs)
        v8f s[4];
        #pragma unroll
        for (int nt = 0; nt < 4; ++nt)
            #pragma unroll
            for (int e = 0; e < 8; ++e) s[nt][e] = 0.0f;

        #pragma unroll
        for (int kk = 0; kk < 2; ++kk) {
            #pragma unroll
            for (int nt = 0; nt < 4; ++nt) {
                FragU bk;
                #pragma unroll
                for (int v = 0; v < 8; ++v)
                    bk.u[v] = Ks[nt * 16 + lrow][KPIDX(kk, v, half)];
                s[nt] = __builtin_amdgcn_wmma_f32_16x16x32_bf16(
                    false, aq[kk].b, false, bk.b, (short)0, s[nt], false, false);
            }
        }

        // scale(+log2e) + causal mask + online softmax in log2 domain
        const int mrow0 = qt * 64 + wave * 16 + 8 * half;   // + v
        const int ncol0 = j * 64 + lrow;                    // + nt*16
        #pragma unroll
        for (int v = 0; v < 8; ++v) {
            float sv[4];
            float rowmax = -__builtin_inff();
            #pragma unroll
            for (int nt = 0; nt < 4; ++nt) {
                float val = s[nt][v] * scale2;
                if (ncol0 + nt * 16 > mrow0 + v) val = -__builtin_inff();
                sv[nt] = val;
                rowmax = fmaxf(rowmax, val);
            }
            #pragma unroll
            for (int off = 1; off < 16; off <<= 1)
                rowmax = fmaxf(rowmax, __shfl_xor(rowmax, off, 32));

            float mnew = fmaxf(mi[v], rowmax);
            float corr = __builtin_amdgcn_exp2f(mi[v] - mnew);   // v_exp_f32
            mi[v] = mnew;
            li[v] *= corr;

            float psum = 0.0f;
            #pragma unroll
            for (int nt = 0; nt < 4; ++nt) {
                float p = __builtin_amdgcn_exp2f(sv[nt] - mnew);
                psum += p;
                ((bf16*)&Ps[wave][v + 8 * half][0])[nt * 16 + lrow] = (bf16)p;
                o[nt][v] *= corr;
            }
            #pragma unroll
            for (int off = 1; off < 16; off <<= 1)
                psum += __shfl_xor(psum, off, 32);
            li[v] += psum;
        }
        __syncthreads();   // Ps C/D-layout store -> A-layout reload

        // O += P @ V   (Vt transposed: pair loads contiguous)
        #pragma unroll
        for (int kk = 0; kk < 2; ++kk) {
            FragU ap;
            #pragma unroll
            for (int v = 0; v < 8; ++v)
                ap.u[v] = Ps[wave][lrow][KPIDX(kk, v, half)];
            #pragma unroll
            for (int nt = 0; nt < 4; ++nt) {
                FragU bv;
                #pragma unroll
                for (int v = 0; v < 8; ++v)
                    bv.u[v] = Vt[nt * 16 + lrow][KPIDX(kk, v, half)];
                o[nt] = __builtin_amdgcn_wmma_f32_16x16x32_bf16(
                    false, ap.b, false, bv.b, (short)0, o[nt], false, false);
            }
        }
    }

    // epilogue: multiply by 1/rowsum (v_rcp_f32, no IEEE div expansion)
    float rli[8];
    #pragma unroll
    for (int v = 0; v < 8; ++v) rli[v] = __builtin_amdgcn_rcpf(li[v]);
    #pragma unroll
    for (int nt = 0; nt < 4; ++nt)
        #pragma unroll
        for (int v = 0; v < 8; ++v) {
            long t = qt * 64 + wave * 16 + v + 8 * half;
            out[base + t * HSN + nt * 16 + lrow] = o[nt][v] * rli[v];
        }
}

// --------------------------------------------------------------------------
extern "C" void kernel_launch(void* const* d_in, const int* in_sizes, int n_in,
                              void* d_out, int out_size, void* d_ws, size_t ws_size,
                              hipStream_t stream) {
    const float* x  = (const float*)d_in[0];
    const float* Wk = (const float*)d_in[1];
    const float* Wq = (const float*)d_in[2];
    const float* Wv = (const float*)d_in[3];
    float* out = (float*)d_out;

    const long nQKV = (long)BN * TN * HSN;          // 8,388,608 elems
    bf16* Kb = (bf16*)d_ws;                         // 3 * nQKV * 2 B = 50.3 MB scratch
    bf16* Qb = Kb + nQKV;
    bf16* Vb = Qb + nQKV;

    dim3 g1((BN * TN) / 64, 3);
    qkv_proj_kernel<<<g1, dim3(128), 0, stream>>>(x, Wk, Wq, Wv, Kb, Qb, Vb);

    dim3 g2(TN / 64, BN);
    flash_attn_kernel<<<g2, dim3(128), 0, stream>>>(Qb, Kb, Vb, out);
}
